// LATTE_53506702574182
// MI455X (gfx1250) — matrix operand order repaired and checked by reading, not compiled, not run on hardware
//
#include <hip/hip_runtime.h>
#include <hip/hip_bf16.h>
#include <stdint.h>

typedef __attribute__((ext_vector_type(16))) _Float16 v16h;
typedef __attribute__((ext_vector_type(4)))  _Float16 v4h;
typedef __attribute__((ext_vector_type(8)))  float    v8f;

#define DIM 128
#define HD  4
#define OCH 32

// ---------- helpers ----------
__device__ __forceinline__ unsigned fkey(float f) {
  // order-preserving float -> unsigned mapping for atomicMax-based segment max
  unsigned u = __float_as_uint(f);
  return (u & 0x80000000u) ? ~u : (u | 0x80000000u);
}
__device__ __forceinline__ float unkey(unsigned k) {
  unsigned u = (k & 0x80000000u) ? (k ^ 0x80000000u) : ~k;
  return __uint_as_float(u);
}
__device__ __forceinline__ float leaky(float v) { return v > 0.f ? v : 0.2f * v; }

__device__ __forceinline__ v8f wmma_f16(const v16h& a, const v16h& b, const v8f& c) {
  return __builtin_amdgcn_wmma_f32_16x16x32_f16(false, a, false, b, (short)0, c,
                                                false, false);
}

// ---------- precision conversion ----------
__global__ void cvt_x_half(const float4* __restrict__ x, v4h* __restrict__ xh, int n4) {
  int i = blockIdx.x * blockDim.x + threadIdx.x;
  if (i >= n4) return;
  float4 v = x[i];
  v4h h;
  h.x = (_Float16)v.x; h.y = (_Float16)v.y; h.z = (_Float16)v.z; h.w = (_Float16)v.w;
  xh[i] = h;
}

// wt[c][k] = (c<128 ? Wl[k][c] : Wr[k][c-128]) as f16; row = output column
__global__ __launch_bounds__(128) void cvt_w_half(const float* __restrict__ Wl,
                                                  const float* __restrict__ Wr,
                                                  _Float16* __restrict__ wt) {
  int c = blockIdx.x;           // 0..255
  int k = threadIdx.x;          // 0..127
  float v = (c < DIM) ? Wl[(size_t)k * DIM + c] : Wr[(size_t)k * DIM + (c - DIM)];
  wt[(size_t)c * DIM + k] = (_Float16)v;
}

// ---------- projection GEMM via WMMA ----------
// grid = (N/16), block = 128 (4 waves). Wave w computes a 16x64 output tile:
// rows [16*bx, +16) x cols [64*w, +64) of the [N,256] concat (l|r).
// 4 accumulators -> 4 independent WMMA chains; A fragment shared across them.
__global__ __launch_bounds__(128) void proj_gemm_wmma(
    const _Float16* __restrict__ xh,   // [N,128]
    const _Float16* __restrict__ wt,   // [256,128], row = out col
    const float* __restrict__ bl, const float* __restrict__ br,
    float* __restrict__ l, float* __restrict__ r) {
  const int lane = threadIdx.x & 31;
  const int wv   = threadIdx.x >> 5;   // 0..3
  const int m0 = blockIdx.x * 16;
  const int n0 = wv * 64;              // 4 col tiles of 16
  const int nr = lane & 15;            // A row / B col / D col
  const int hs = lane >> 4;            // lane half-select
  const _Float16* arow = xh + (size_t)(m0 + nr) * DIM;
  const _Float16* brow = wt + (size_t)(n0 + nr) * DIM;
  v8f acc0 = {}, acc1 = {}, acc2 = {}, acc3 = {};
#pragma unroll
  for (int k0 = 0; k0 < DIM; k0 += 32) {
    v16h a, b0, b1, b2, b3;
#pragma unroll
    for (int j = 0; j < 8; ++j) {
      // A k order used here (contiguous, not the ISA table order): lanes 0-15 K=0..15, lanes 16-31 K=16..31
      const int ka = 2 * j + 16 * hs;
      // B k order used here (same contiguous order as A): lanes 0-15 K=0..15, lanes 16-31 K=16..31
      const int kb = 2 * j + 16 * hs;
      a[2 * j]      = arow[k0 + ka];
      a[2 * j + 1]  = arow[k0 + ka + 1];
      b0[2 * j]     = brow[k0 + kb];
      b0[2 * j + 1] = brow[k0 + kb + 1];
      b1[2 * j]     = brow[16 * DIM + k0 + kb];
      b1[2 * j + 1] = brow[16 * DIM + k0 + kb + 1];
      b2[2 * j]     = brow[32 * DIM + k0 + kb];
      b2[2 * j + 1] = brow[32 * DIM + k0 + kb + 1];
      b3[2 * j]     = brow[48 * DIM + k0 + kb];
      b3[2 * j + 1] = brow[48 * DIM + k0 + kb + 1];
    }
    acc0 = wmma_f16(a, b0, acc0);
    acc1 = wmma_f16(a, b1, acc1);
    acc2 = wmma_f16(a, b2, acc2);
    acc3 = wmma_f16(a, b3, acc3);
  }
  // D layout: VGPR v, lane L -> row m0 + v + 8*(L/16), col tile base + (L%16)
  v8f accs[4] = {acc0, acc1, acc2, acc3};
#pragma unroll
  for (int t = 0; t < 4; ++t) {
    const int nc = n0 + t * 16 + nr;
#pragma unroll
    for (int v = 0; v < 8; ++v) {
      const int m = m0 + v + 8 * hs;
      float val = accs[t][v];
      if (nc < DIM) l[(size_t)m * DIM + nc] = val + bl[nc];
      else          r[(size_t)m * DIM + (nc - DIM)] = val + br[nc - DIM];
    }
  }
}

// ---------- per-node scores + beta softmax ----------
// 1 block = 1 node, 128 threads (wave per head for the score reductions)
__global__ __launch_bounds__(128) void node_scores(
    const float* __restrict__ x, const float* __restrict__ Wc, const float* __restrict__ bc,
    const float* __restrict__ l, const float* __restrict__ r,
    const float* __restrict__ attn1, const float* __restrict__ attn2,
    const float* __restrict__ sharp,
    float* __restrict__ beta,
    float* __restrict__ si1, float* __restrict__ sj1,
    float* __restrict__ si2, float* __restrict__ sj2) {
  const int n = blockIdx.x;
  const int tid = threadIdx.x;
  __shared__ float xs[DIM];
  __shared__ float blog[HD * 3];
  xs[tid] = x[(size_t)n * DIM + tid];
  __syncthreads();
  if (tid < HD * 3) {            // beta logits: x @ Wc.T + bc, Wc = [12,128]
    float acc = bc[tid];
    const float* wrow = Wc + (size_t)tid * DIM;
#pragma unroll 4
    for (int k = 0; k < DIM; ++k) acc += xs[k] * wrow[k];
    blog[tid] = acc;
  }
  __syncthreads();
  if (tid < HD) {                // softmax over R=3 per head
    float a0 = blog[tid * 3 + 0], a1 = blog[tid * 3 + 1], a2 = blog[tid * 3 + 2];
    float mx = fmaxf(a0, fmaxf(a1, a2));
    float e0 = __expf(a0 - mx), e1 = __expf(a1 - mx), e2 = __expf(a2 - mx);
    float inv = 1.0f / (e0 + e1 + e2);
    beta[(size_t)n * 12 + tid * 3 + 0] = e0 * inv;
    beta[(size_t)n * 12 + tid * 3 + 1] = e1 * inv;
    beta[(size_t)n * 12 + tid * 3 + 2] = e2 * inv;
  }
  // attention score halves: wave h handles head h, lane c = channel
  const int h = tid >> 5;
  const int c = tid & 31;
  const float lv = l[(size_t)n * DIM + h * OCH + c];
  const float rv = r[(size_t)n * DIM + h * OCH + c];
  const float llv = leaky(lv), lrv = leaky(rv);
  float p_i1 = lrv * attn1[h * 2 * OCH + c];
  float p_j1 = llv * attn1[h * 2 * OCH + OCH + c];
  float p_i2 = lrv * attn2[h * 2 * OCH + c];
  float p_j2 = llv * attn2[h * 2 * OCH + OCH + c];
#pragma unroll
  for (int off = 16; off > 0; off >>= 1) {   // wave32 tree reduction
    p_i1 += __shfl_down(p_i1, off, 32);
    p_j1 += __shfl_down(p_j1, off, 32);
    p_i2 += __shfl_down(p_i2, off, 32);
    p_j2 += __shfl_down(p_j2, off, 32);
  }
  if (c == 0) {
    const float s1 = sharp[0], s2 = sharp[1];
    si1[n * HD + h] = s1 * p_i1;
    sj1[n * HD + h] = s1 * p_j1;
    si2[n * HD + h] = s2 * p_i2;
    sj2[n * HD + h] = s2 * p_j2;
  }
}

// ---------- scratch init ----------
__global__ void fill_u32(unsigned* __restrict__ p, unsigned v, long long n) {
  long long i = (long long)blockIdx.x * blockDim.x + threadIdx.x;
  if (i < n) p[i] = v;
}

// ---------- edge passes (per relation) ----------
__global__ void edge_pass_max(const int* __restrict__ ei,
                              const float* __restrict__ si, const float* __restrict__ sj,
                              unsigned* __restrict__ mkey, int E) {
  int e = blockIdx.x * blockDim.x + threadIdx.x;
  if (e >= E) return;
  const int s = ei[e];
  const int d = ei[E + e];
#pragma unroll
  for (int h = 0; h < HD; ++h) {
    float logit = si[d * HD + h] + sj[s * HD + h];
    atomicMax(&mkey[d * HD + h], fkey(logit));
  }
}

__global__ void edge_pass_sum(const int* __restrict__ ei,
                              const float* __restrict__ si, const float* __restrict__ sj,
                              const unsigned* __restrict__ mkey,
                              float* __restrict__ denom, int E) {
  int e = blockIdx.x * blockDim.x + threadIdx.x;
  if (e >= E) return;
  const int s = ei[e];
  const int d = ei[E + e];
#pragma unroll
  for (int h = 0; h < HD; ++h) {
    float logit = si[d * HD + h] + sj[s * HD + h];
    atomicAdd(&denom[d * HD + h], __expf(logit - unkey(mkey[d * HD + h])));
  }
}

// wave32 per edge: lane c scatters channel c of each head's 32-wide message
__global__ void edge_pass_scatter(const int* __restrict__ ei,
                                  const float* __restrict__ si, const float* __restrict__ sj,
                                  const unsigned* __restrict__ mkey,
                                  const float* __restrict__ denom,
                                  const float* __restrict__ l,
                                  float* __restrict__ agg, int E) {
  long long gid = (long long)blockIdx.x * blockDim.x + threadIdx.x;
  int e = (int)(gid >> 5);
  int c = (int)(gid & 31);
  if (e >= E) return;
  const int s = ei[e];
  const int d = ei[E + e];
  const float* lrow = l + (size_t)s * DIM;
  float* arow = agg + (size_t)d * DIM;
#pragma unroll
  for (int h = 0; h < HD; ++h) {
    float logit = si[d * HD + h] + sj[s * HD + h];
    float alpha = __expf(logit - unkey(mkey[d * HD + h])) / denom[d * HD + h];
    atomicAdd(&arow[h * OCH + c], lrow[h * OCH + c] * alpha);
  }
}

// ---------- beta-weighted fusion + relu ----------
__global__ void fuse_relu(const float* __restrict__ agg1, const float* __restrict__ agg2,
                          const float* __restrict__ r, const float* __restrict__ beta,
                          float* __restrict__ out, long long total) {
  long long i = (long long)blockIdx.x * blockDim.x + threadIdx.x;
  if (i >= total) return;
  int n = (int)(i >> 7);
  int d = (int)(i & 127);
  int h = d >> 5;
  const float* b = beta + (size_t)n * 12 + h * 3;
  float v = b[0] * agg1[i] + b[1] * agg2[i] + b[2] * r[i];
  out[i] = fmaxf(v, 0.f);
}

// ---------- launcher ----------
extern "C" void kernel_launch(void* const* d_in, const int* in_sizes, int n_in,
                              void* d_out, int out_size, void* d_ws, size_t ws_size,
                              hipStream_t stream) {
  const float* x     = (const float*)d_in[0];
  const float* Wl    = (const float*)d_in[1];
  const float* bl    = (const float*)d_in[2];
  const float* Wr    = (const float*)d_in[3];
  const float* br    = (const float*)d_in[4];
  const float* Wc    = (const float*)d_in[5];
  const float* bc    = (const float*)d_in[6];
  const float* attn1 = (const float*)d_in[7];
  const float* attn2 = (const float*)d_in[8];
  const float* sharp = (const float*)d_in[9];
  const int*   ei1   = (const int*)d_in[10];
  const int*   ei2   = (const int*)d_in[11];

  const int N = in_sizes[0] / DIM;     // 100000 (multiple of 16)
  const int E = in_sizes[10] / 2;      // 1600000

  // workspace carve-out (256B aligned slices)
  char* ws = (char*)d_ws;
  size_t off = 0;
  auto carve = [&](size_t bytes) -> char* {
    char* p = ws + off;
    off += (bytes + 255) & ~(size_t)255;
    return p;
  };
  _Float16* xh  = (_Float16*)carve((size_t)N * DIM * sizeof(_Float16));
  _Float16* wt  = (_Float16*)carve((size_t)2 * DIM * DIM * sizeof(_Float16));
  float* l      = (float*)carve((size_t)N * DIM * sizeof(float));
  float* r      = (float*)carve((size_t)N * DIM * sizeof(float));
  float* beta   = (float*)carve((size_t)N * 12 * sizeof(float));
  float* si1    = (float*)carve((size_t)N * HD * sizeof(float));
  float* sj1    = (float*)carve((size_t)N * HD * sizeof(float));
  float* si2    = (float*)carve((size_t)N * HD * sizeof(float));
  float* sj2    = (float*)carve((size_t)N * HD * sizeof(float));
  // zero-init region: [mkey1 .. agg2]
  char* zbase   = ws + off;
  unsigned* mk1 = (unsigned*)carve((size_t)N * HD * sizeof(unsigned));
  unsigned* mk2 = (unsigned*)carve((size_t)N * HD * sizeof(unsigned));
  float* den1   = (float*)carve((size_t)N * HD * sizeof(float));
  float* den2   = (float*)carve((size_t)N * HD * sizeof(float));
  float* agg1   = (float*)carve((size_t)N * DIM * sizeof(float));
  float* agg2   = (float*)carve((size_t)N * DIM * sizeof(float));
  const long long zcount = (long long)((ws + off) - zbase) / 4;

  const int BT = 256;

  // 1) precision conversion
  {
    int n4 = (N * DIM) / 4;
    cvt_x_half<<<(n4 + BT - 1) / BT, BT, 0, stream>>>((const float4*)x, (v4h*)xh, n4);
    cvt_w_half<<<2 * DIM, DIM, 0, stream>>>(Wl, Wr, wt);
  }
  // 2) WMMA projection GEMM: [N,128] x [128,256] -> l | r
  proj_gemm_wmma<<<N / 16, 128, 0, stream>>>(xh, wt, bl, br, l, r);
  // 3) per-node scores + beta
  node_scores<<<N, DIM, 0, stream>>>(x, Wc, bc, l, r, attn1, attn2, sharp,
                                     beta, si1, sj1, si2, sj2);
  // 4) zero segment-max keys / denominators / aggregation buffers
  fill_u32<<<(unsigned)((zcount + BT - 1) / BT), BT, 0, stream>>>((unsigned*)zbase, 0u, zcount);
  // 5) edge softmax + scatter, relation 1 then relation 2
  {
    int gE = (E + BT - 1) / BT;
    long long st = (long long)E * 32;
    unsigned gS = (unsigned)((st + BT - 1) / BT);
    edge_pass_max<<<gE, BT, 0, stream>>>(ei1, si1, sj1, mk1, E);
    edge_pass_max<<<gE, BT, 0, stream>>>(ei2, si2, sj2, mk2, E);
    edge_pass_sum<<<gE, BT, 0, stream>>>(ei1, si1, sj1, mk1, den1, E);
    edge_pass_sum<<<gE, BT, 0, stream>>>(ei2, si2, sj2, mk2, den2, E);
    edge_pass_scatter<<<gS, BT, 0, stream>>>(ei1, si1, sj1, mk1, den1, l, agg1, E);
    edge_pass_scatter<<<gS, BT, 0, stream>>>(ei2, si2, sj2, mk2, den2, l, agg2, E);
  }
  // 6) fusion + relu -> d_out
  {
    long long total = (long long)N * DIM;
    fuse_relu<<<(unsigned)((total + BT - 1) / BT), BT, 0, stream>>>(
        agg1, agg2, r, beta, (float*)d_out, total);
  }
}
